// ScannedRNN_24816321036373
// MI455X (gfx1250) — matrix operand order repaired and checked by reading, not compiled, not run on hardware
//
#include <hip/hip_runtime.h>
#include <stdint.h>

// Problem constants (from the reference)
#define T_STEPS 512
#define B_SZ    256
#define H_SZ    512
#define KT_N    (H_SZ / 32)   // 16 K-tiles of 32
#define NT_N    (H_SZ / 16)   // 32 N-tiles of 16
#define MT_N    (B_SZ / 16)   // 16 M-tiles of 16
#define HH      (H_SZ * H_SZ)
#define BH      (B_SZ * H_SZ)

typedef __attribute__((ext_vector_type(16))) __bf16       v16bf;
typedef __attribute__((ext_vector_type(8)))  float        v8f;
typedef __attribute__((ext_vector_type(8)))  unsigned int u32x8;
typedef __attribute__((ext_vector_type(4)))  unsigned int nu4;

// ---------------------------------------------------------------------------
// fp32 -> bf16 (RNE) bit math — used ONLY in one-time shuffles and epilogue
// ---------------------------------------------------------------------------
__device__ __forceinline__ unsigned int rnebf(float f) {
  unsigned int u = __float_as_uint(f);
  return (u + 0x7FFFu + ((u >> 16) & 1u)) >> 16;
}

// ---------------------------------------------------------------------------
// WMMA wrapper: D = A(16x32 bf16) * B(32x16 bf16) + C(16x16 f32)
// ---------------------------------------------------------------------------
__device__ __forceinline__ v8f wmma_bf16(v16bf a, v16bf b, v8f c) {
  return __builtin_amdgcn_wmma_f32_16x16x32_bf16(
      false, a, false, b, (short)0, c, false, false);
}

// ---------------------------------------------------------------------------
// Fragment loaders: every fragment is one contiguous 32B chunk per lane.
// ---------------------------------------------------------------------------
__device__ __forceinline__ v16bf load_frag(const unsigned short* __restrict__ base,
                                           int fragIdx, int lane) {
  const nu4* p = reinterpret_cast<const nu4*>(base + (size_t)((fragIdx * 32 + lane) << 4));
  nu4 x = p[0];
  nu4 y = p[1];
  u32x8 u;
  u[0] = x[0]; u[1] = x[1]; u[2] = x[2]; u[3] = x[3];
  u[4] = y[0]; u[5] = y[1]; u[6] = y[2]; u[7] = y[3];
  return __builtin_bit_cast(v16bf, u);
}

// Non-temporal variant for the single-use ins fragments (don't pollute L2)
__device__ __forceinline__ v16bf load_frag_nt(const unsigned short* __restrict__ base,
                                              int fragIdx, int lane) {
  const nu4* p = reinterpret_cast<const nu4*>(base + (size_t)((fragIdx * 32 + lane) << 4));
  nu4 x = __builtin_nontemporal_load(p);
  nu4 y = __builtin_nontemporal_load(p + 1);
  u32x8 u;
  u[0] = x[0]; u[1] = x[1]; u[2] = x[2]; u[3] = x[3];
  u[4] = y[0]; u[5] = y[1]; u[6] = y[2]; u[7] = y[3];
  return __builtin_bit_cast(v16bf, u);
}

// Apply reset mask to an already-loaded fragment (8 v_and/cndmask ops).
__device__ __forceinline__ v16bf mask_frag(v16bf a, unsigned int maskbits) {
  u32x8 u = __builtin_bit_cast(u32x8, a);
  u[0] &= maskbits; u[1] &= maskbits; u[2] &= maskbits; u[3] &= maskbits;
  u[4] &= maskbits; u[5] &= maskbits; u[6] &= maskbits; u[7] &= maskbits;
  return __builtin_bit_cast(v16bf, u);
}

__device__ __forceinline__ float sigmoidf_(float x) {
  return 1.0f / (1.0f + __expf(-x));
}

// A-layout index helpers (16-bit A 16x32 per CDNA5 ISA):
//   lane<16 : row M=lane,    K = kt*32 + {e&7 for e<8 ; 16+(e&7) for e>=8}
//   lane>=16: row M=lane-16, K offsets shifted by +8
// For a given hidden index k within a 32-wide K-tile (r = k%32):
//   inHi = (r>>3)&1 ; e = (r&7) + ((r>>4)<<3) ; laneA = (row%16) + (inHi<<4)

// ---------------------------------------------------------------------------
// One-time weight shuffle: fp32 W[H][H] (k-major) -> bf16 WMMA-B layout.
// B layout: lane<16: K = kt*32 + {e&7 for e<8 ; 16+(e&7) for e>=8}, N = nt*16+lane; lane>=16: K += 8, N -= 16
// ---------------------------------------------------------------------------
__global__ __launch_bounds__(256) void gru_shuffle_weights(
    const float* __restrict__ W0, const float* __restrict__ W1,
    const float* __restrict__ W2, const float* __restrict__ W3,
    const float* __restrict__ W4, const float* __restrict__ W5,
    unsigned short* __restrict__ out) {
  int i = blockIdx.x * blockDim.x + threadIdx.x;
  if (i >= 6 * HH) return;
  int e     = i & 15;
  int l     = (i >> 4) & 31;
  int rest  = i >> 9;
  int kt    = rest & (KT_N - 1);
  int rest2 = rest >> 4;
  int nt    = rest2 & (NT_N - 1);
  int w     = rest2 >> 5;
  int k = kt * 32 + ((l >> 4) << 3) + (e & 7) + ((e >> 3) << 4);
  int n = nt * 16 + (l & 15);
  const float* W = (w == 0) ? W0 : (w == 1) ? W1 : (w == 2) ? W2
                 : (w == 3) ? W3 : (w == 4) ? W4 : W5;
  out[i] = (unsigned short)rnebf(W[(size_t)k * H_SZ + n]);
}

// ---------------------------------------------------------------------------
// One-time ins shuffle: fp32 ins[T][B][H] -> bf16 A-fragment layout
// [t][mt][kt][lane][e]  (2^17 elements per t)
// ---------------------------------------------------------------------------
__global__ __launch_bounds__(256) void gru_shuffle_ins(
    const float* __restrict__ ins, unsigned short* __restrict__ out, int total) {
  int i = blockIdx.x * blockDim.x + threadIdx.x;
  if (i >= total) return;
  int e    = i & 15;
  int lane = (i >> 4) & 31;
  int kt   = (i >> 9) & (KT_N - 1);
  int mt   = (i >> 13) & (MT_N - 1);
  int t    = i >> 17;
  int row  = mt * 16 + (lane & 15);
  int k    = kt * 32 + ((lane >> 4) << 3) + (e & 7) + ((e >> 3) << 4);
  out[i] = (unsigned short)rnebf(ins[(size_t)t * BH + (size_t)row * H_SZ + k]);
}

// Same mapping for the initial carry h0 (single timestep) -> hA[0]
__global__ __launch_bounds__(256) void gru_shuffle_h0(
    const float* __restrict__ h0, unsigned short* __restrict__ out) {
  int i = blockIdx.x * blockDim.x + threadIdx.x;
  if (i >= BH) return;
  int e    = i & 15;
  int lane = (i >> 4) & 31;
  int kt   = (i >> 9) & (KT_N - 1);
  int mt   = (i >> 13) & (MT_N - 1);
  int row  = mt * 16 + (lane & 15);
  int k    = kt * 32 + ((lane >> 4) << 3) + (e & 7) + ((e >> 3) << 4);
  out[i] = (unsigned short)rnebf(h0[(size_t)row * H_SZ + k]);
}

// ---------------------------------------------------------------------------
// One fused GRU step. Each wave computes TWO 16x16 output tiles stacked in M
// (rows m0..m0+31, cols nt*16..+15): per K-tile it loads 4 A + 6 B fragments
// and issues 12 WMMAs across 8 accumulator chains (0.83 loads/WMMA), giving
// the scheduler enough independent XDL work to cover b128 load latency.
// Grid: 256 wave-tiles / 8 waves -> 32 blocks of 256 threads.
// ---------------------------------------------------------------------------
__global__ __launch_bounds__(256) void gru_step_kernel(
    const unsigned short* __restrict__ insA,   // bf16 A-layout slice at t
    const unsigned short* __restrict__ hAprev, // bf16 A-layout h_{t-1}
    unsigned short* __restrict__ hAnext,       // bf16 A-layout h_t (written)
    const float* __restrict__ hprevF,          // fp32 h_{t-1} (d_out[t-1]/h0)
    const unsigned char* __restrict__ resetsT, // [B] bool slice at t
    const unsigned short* __restrict__ wsh,    // 6 shuffled bf16 weights
    const float* __restrict__ bir, const float* __restrict__ biz,
    const float* __restrict__ bin_, const float* __restrict__ bhn,
    float* __restrict__ outT) {                // fp32 [B][H] slice at t
  const int lane = threadIdx.x & 31;
  const int wid  = threadIdx.x >> 5;
  const int wt   = blockIdx.x * 8 + wid;       // 0..255 wave-tiles
  const int mp   = wt >> 5;                    // m-pair 0..7
  const int nt   = wt & (NT_N - 1);            // n-tile 0..31
  const int mt0  = mp * 2;
  const int mt1  = mp * 2 + 1;

  const unsigned short* Wir_s = wsh + (size_t)0 * HH;
  const unsigned short* Wiz_s = wsh + (size_t)1 * HH;
  const unsigned short* Win_s = wsh + (size_t)2 * HH;
  const unsigned short* Whr_s = wsh + (size_t)3 * HH;
  const unsigned short* Whz_s = wsh + (size_t)4 * HH;
  const unsigned short* Whn_s = wsh + (size_t)5 * HH;

  v8f accR0 = {}; v8f accZ0 = {}; v8f accP0 = {}; v8f accQ0 = {};
  v8f accR1 = {}; v8f accZ1 = {}; v8f accP1 = {}; v8f accQ1 = {};

  const int rowA0 = mt0 * 16 + (lane & 15);
  const int rowA1 = mt1 * 16 + (lane & 15);
  const unsigned int hmask0 = resetsT[rowA0] ? 0u : 0xFFFFFFFFu;
  const unsigned int hmask1 = resetsT[rowA1] ? 0u : 0xFFFFFFFFu;

  #pragma unroll
  for (int kt = 0; kt < KT_N; ++kt) {
    const int fa0 = mt0 * KT_N + kt;
    const int fa1 = mt1 * KT_N + kt;
    const int fb  = nt * KT_N + kt;
    v16bf aI0 = load_frag_nt(insA, fa0, lane);
    v16bf aI1 = load_frag_nt(insA, fa1, lane);
    v16bf aH0 = mask_frag(load_frag(hAprev, fa0, lane), hmask0);
    v16bf aH1 = mask_frag(load_frag(hAprev, fa1, lane), hmask1);
    v16bf bIr = load_frag(Wir_s, fb, lane);
    v16bf bIz = load_frag(Wiz_s, fb, lane);
    v16bf bIn = load_frag(Win_s, fb, lane);
    v16bf bHr = load_frag(Whr_s, fb, lane);
    v16bf bHz = load_frag(Whz_s, fb, lane);
    v16bf bHn = load_frag(Whn_s, fb, lane);
    accR0 = wmma_bf16(aI0, bIr, accR0);
    accR1 = wmma_bf16(aI1, bIr, accR1);
    accZ0 = wmma_bf16(aI0, bIz, accZ0);
    accZ1 = wmma_bf16(aI1, bIz, accZ1);
    accP0 = wmma_bf16(aI0, bIn, accP0);
    accP1 = wmma_bf16(aI1, bIn, accP1);
    accR0 = wmma_bf16(aH0, bHr, accR0);
    accR1 = wmma_bf16(aH1, bHr, accR1);
    accZ0 = wmma_bf16(aH0, bHz, accZ0);
    accZ1 = wmma_bf16(aH1, bHz, accZ1);
    accQ0 = wmma_bf16(aH0, bHn, accQ0);
    accQ1 = wmma_bf16(aH1, bHn, accQ1);
  }

  // ---- Epilogue: gates + blend for both tiles; C/D layout:
  //      row = m0 + i + 8*(lane>=16), col = nt*16 + lane%16.
  //      Also scatter h_new as bf16 into hAnext (A-fragment layout). ----
  const int col = nt * 16 + (lane & 15);
  const float bR = bir[col];
  const float bZ = biz[col];
  const float bN = bin_[col];
  const float bH = bhn[col];

  // A-layout coordinates of hidden index `col` for the hAnext scatter
  const int ktH  = col >> 5;
  const int r32  = col & 31;
  const int inHi = (r32 >> 3) & 1;
  const int eA   = (r32 & 7) + ((r32 >> 4) << 3);

  #pragma unroll
  for (int tt = 0; tt < 2; ++tt) {
    const int mt = (tt == 0) ? mt0 : mt1;
    const v8f aR = (tt == 0) ? accR0 : accR1;
    const v8f aZ = (tt == 0) ? accZ0 : accZ1;
    const v8f aP = (tt == 0) ? accP0 : accP1;
    const v8f aQ = (tt == 0) ? accQ0 : accQ1;
    const int rowBase = mt * 16 + ((lane >> 4) << 3);
    #pragma unroll
    for (int i = 0; i < 8; ++i) {
      const int row = rowBase + i;
      const float hm = resetsT[row] ? 0.0f : hprevF[(size_t)row * H_SZ + col];
      const float r  = sigmoidf_(aR[i] + bR);
      const float z  = sigmoidf_(aZ[i] + bZ);
      const float n  = tanhf(aP[i] + bN + r * (aQ[i] + bH));
      const float hn = (1.0f - z) * n + z * hm;
      outT[(size_t)row * H_SZ + col] = hn;
      const int laneA = (row & 15) + (inHi << 4);
      hAnext[(size_t)(((mt * KT_N + ktH) * 32 + laneA) << 4) + eA] =
          (unsigned short)rnebf(hn);
    }
  }
}

// ---------------------------------------------------------------------------
// Host-side launch. Workspace layout (bytes):
//   [0, 3MB)              : 6 shuffled bf16 weights (L2-resident all run)
//   [3MB, 3MB+134MB)      : ins in bf16 A-fragment layout
//   then 2 x 256KB        : bf16 h ping-pong (A-fragment layout)
// fp32 hidden state chains through d_out itself. All launches on `stream`.
// ---------------------------------------------------------------------------
extern "C" void kernel_launch(void* const* d_in, const int* in_sizes, int n_in,
                              void* d_out, int out_size, void* d_ws, size_t ws_size,
                              hipStream_t stream) {
  const float*         ins    = (const float*)d_in[0];
  const unsigned char* resets = (const unsigned char*)d_in[1];
  const float*         h0     = (const float*)d_in[2];
  const float*         Wir    = (const float*)d_in[3];
  const float*         Wiz    = (const float*)d_in[4];
  const float*         Win    = (const float*)d_in[5];
  const float*         Whr    = (const float*)d_in[6];
  const float*         Whz    = (const float*)d_in[7];
  const float*         Whn    = (const float*)d_in[8];
  const float*         bir    = (const float*)d_in[9];
  const float*         biz    = (const float*)d_in[10];
  const float*         bin_   = (const float*)d_in[11];
  const float*         bhn    = (const float*)d_in[12];
  float*               out    = (float*)d_out;

  unsigned short* wsh  = (unsigned short*)d_ws;
  unsigned short* insA = wsh + (size_t)6 * HH;                       // +3 MB
  unsigned short* hA0  = insA + (size_t)T_STEPS * BH;                // +134 MB
  unsigned short* hA1  = hA0 + (size_t)BH;

  // 1) one-time shuffles: weights, ins, h0 (all bf16, WMMA-operand layout)
  gru_shuffle_weights<<<(6 * HH + 255) / 256, 256, 0, stream>>>(
      Wir, Wiz, Win, Whr, Whz, Whn, wsh);
  {
    const int total = T_STEPS * BH;
    gru_shuffle_ins<<<(total + 255) / 256, 256, 0, stream>>>(ins, insA, total);
  }
  gru_shuffle_h0<<<(BH + 255) / 256, 256, 0, stream>>>(h0, hA0);

  // 2) sequential scan: 512 fused step kernels; bf16 state ping-pongs in ws,
  //    fp32 state chains through d_out
  unsigned short* hA[2] = {hA0, hA1};
  for (int t = 0; t < T_STEPS; ++t) {
    const float* hprevF = (t == 0) ? h0 : (out + (size_t)(t - 1) * BH);
    gru_step_kernel<<<(MT_N * NT_N) / 16, 256, 0, stream>>>(
        insA + (size_t)t * BH, hA[t & 1], hA[(t + 1) & 1], hprevF,
        resets + (size_t)t * B_SZ, wsh,
        bir, biz, bin_, bhn, out + (size_t)t * BH);
  }
}
